// AttentionLanguageModel_58110907515491
// MI455X (gfx1250) — matrix operand order repaired and checked by reading, not compiled, not run on hardware
//
#include <hip/hip_runtime.h>

// ---------------- dims ----------------
constexpr int Bb = 512, Tt = 128, Cc = 96, Hh = 6, HSs = 16, Ll = 6, Vv = 65, FFf = 384;
constexpr int MM = Bb * Tt;          // 65536 tokens
constexpr int VPAD = 80;             // LM-head N padded to multiple of 16

typedef __attribute__((ext_vector_type(16))) _Float16 v16h;
typedef __attribute__((ext_vector_type(8)))  _Float16 v8h;
typedef __attribute__((ext_vector_type(8)))  float    v8f;

union FragH { v16h v; v8h h[2]; _Float16 e[16]; };

// ---------------- weight convert / transpose (f32 -> f16, row=n, col=k) ----------------
// Wq/Wk/Wv: (L,H,C,HS) -> per-layer [96][96] with out row = h*HS+d, col = c
__global__ void cvt_qkv_kernel(const float* __restrict__ in, _Float16* __restrict__ out) {
  int t = blockIdx.x * 256 + threadIdx.x;            // L*H*C*HS = 55296
  int d = t & 15;
  int c = (t >> 4) % Cc;
  int h = ((t >> 4) / Cc) % Hh;
  int l = t / (16 * Cc * Hh);
  out[(size_t)l * Cc * Cc + (h * HSs + d) * Cc + c] = (_Float16)in[t];
}
// Wp: (L,C,C) row-major [cin][cout] -> [n][c]
__global__ void cvt_wp_kernel(const float* __restrict__ in, _Float16* __restrict__ out) {
  int t = blockIdx.x * 256 + threadIdx.x;            // 55296
  int n = t % Cc;
  int c = (t / Cc) % Cc;
  int l = t / (Cc * Cc);
  out[(size_t)l * Cc * Cc + n * Cc + c] = (_Float16)in[t];
}
// W1: (L,C,FF) -> [n(FF)][c]
__global__ void cvt_w1_kernel(const float* __restrict__ in, _Float16* __restrict__ out) {
  int t = blockIdx.x * 256 + threadIdx.x;            // 6*96*384 = 221184
  int n = t % FFf;
  int c = (t / FFf) % Cc;
  int l = t / (Cc * FFf);
  out[(size_t)l * Cc * FFf + n * Cc + c] = (_Float16)in[t];
}
// W2: (L,FF,C) -> [n(C)][f]
__global__ void cvt_w2_kernel(const float* __restrict__ in, _Float16* __restrict__ out) {
  int t = blockIdx.x * 256 + threadIdx.x;            // 221184
  int n = t % Cc;
  int f = (t / Cc) % FFf;
  int l = t / (Cc * FFf);
  out[(size_t)l * Cc * FFf + n * FFf + f] = (_Float16)in[t];
}
// Wlm: (C,V) -> [n(0..79)][c], zero-padded rows >= 65
__global__ void cvt_lm_kernel(const float* __restrict__ in, _Float16* __restrict__ out) {
  int t = blockIdx.x * 256 + threadIdx.x;            // 80*96 = 7680
  int c = t % Cc;
  int n = t / Cc;
  out[(size_t)n * Cc + c] = (n < Vv) ? (_Float16)in[c * Vv + n] : (_Float16)0.0f;
}

// ---------------- embedding ----------------
__global__ void embed_kernel(const int* __restrict__ idx, const float* __restrict__ tok,
                             const float* __restrict__ pos, float* __restrict__ x) {
  size_t i = (size_t)blockIdx.x * 256 + threadIdx.x;  // B*T*C
  int c = (int)(i % Cc);
  size_t bt = i / Cc;
  int t = (int)(bt % Tt);
  x[i] = tok[(size_t)idx[bt] * Cc + c] + pos[t * Cc + c];
}

// ---------------- LayerNorm (one wave32 per token), f32 in -> f16 out ----------------
__global__ __launch_bounds__(256) void ln_kernel(const float* __restrict__ x,
                                                 const float* __restrict__ g,
                                                 const float* __restrict__ bta,
                                                 _Float16* __restrict__ out) {
  int tok  = (blockIdx.x << 3) + (threadIdx.x >> 5);
  int lane = threadIdx.x & 31;
  const float* row = x + (size_t)tok * Cc;
  float a0 = row[lane], a1 = row[lane + 32], a2 = row[lane + 64];
  float s = a0 + a1 + a2;
#pragma unroll
  for (int m = 16; m >= 1; m >>= 1) s += __shfl_xor(s, m, 32);
  float mean = s * (1.0f / 96.0f);
  float d0 = a0 - mean, d1 = a1 - mean, d2 = a2 - mean;
  float q = d0 * d0 + d1 * d1 + d2 * d2;
#pragma unroll
  for (int m = 16; m >= 1; m >>= 1) q += __shfl_xor(q, m, 32);
  float rs = rsqrtf(q * (1.0f / 96.0f) + 1e-5f);
  _Float16* orow = out + (size_t)tok * Cc;
  orow[lane]      = (_Float16)(d0 * rs * g[lane]      + bta[lane]);
  orow[lane + 32] = (_Float16)(d1 * rs * g[lane + 32] + bta[lane + 32]);
  orow[lane + 64] = (_Float16)(d2 * rs * g[lane + 64] + bta[lane + 64]);
}

// ---------------- strip-mined WMMA GEMM ----------------
// Y[M,N] = A[M,K](f16) x Bt[N,K](f16); one wave computes a 16 x (16*NT) strip:
// A fragment loaded once per K-step, NT WMMAs against it (amortizes A traffic).
// Epilogue: +bias, +resid (f32), optional ReLU, store f32 and/or f16.
template <int NT>
__global__ __launch_bounds__(256) void gemm_f16_strip(
    const _Float16* __restrict__ A, const _Float16* __restrict__ Bt,
    const float* __restrict__ bias, const float* resid,
    float* outF, _Float16* __restrict__ outH,
    int K, int Npad, int Nreal, int relu) {
  int wave = (blockIdx.x << 3) + (threadIdx.x >> 5);
  int lane = threadIdx.x & 31;
  int strips = (Npad >> 4) / NT;
  int mtile  = wave / strips;
  int ntile0 = (wave % strips) * NT;
  int hs = lane >> 4, l16 = lane & 15;

  const _Float16* arow = A  + (size_t)(mtile * 16 + l16) * K;
  const _Float16* brow = Bt + (size_t)(ntile0 * 16 + l16) * K;

  v8f acc[NT];
  v8f zf = {0.f, 0.f, 0.f, 0.f, 0.f, 0.f, 0.f, 0.f};
#pragma unroll
  for (int s = 0; s < NT; ++s) acc[s] = zf;

  for (int k0 = 0; k0 < K; k0 += 32) {
    FragH a;
    a.v = *(const v16h*)(arow + k0 + hs * 16);        // A: k = hs*16 + e
#pragma unroll
    for (int s = 0; s < NT; ++s) {
      v16h b = *(const v16h*)(brow + (size_t)s * 16 * K + k0 + hs * 16); // B: k = hs*16 + e
      acc[s] = __builtin_amdgcn_wmma_f32_16x16x32_f16(false, a.v, false, b,
                                                      (short)0, acc[s], false, false);
    }
  }

#pragma unroll
  for (int s = 0; s < NT; ++s) {
    int nglob = (ntile0 + s) * 16 + l16;
    bool nok = nglob < Nreal;
    float bi = (bias && nok) ? bias[nglob] : 0.0f;
#pragma unroll
    for (int r = 0; r < 8; ++r) {
      int mglob = mtile * 16 + r + 8 * hs;            // D: VGPR r -> M = r + 8*(lane/16)
      float v = acc[s][r] + bi;
      if (resid && nok) v += resid[(size_t)mglob * Nreal + nglob];
      if (relu) v = fmaxf(v, 0.0f);
      if (nok) {
        if (outF) outF[(size_t)mglob * Nreal + nglob] = v;
        if (outH) outH[(size_t)mglob * Nreal + nglob] = (_Float16)v;
      }
    }
  }
}

// ---------------- fused causal attention, one block per (b,h), 8 waves ----------------
// Wave w owns query rows [16w, 16w+16). Causality: S tiles j > w are fully masked
// (skipped), and P.V K-steps j2 > w/2 contribute nothing (P == 0 there).
__global__ __launch_bounds__(256) void attn_kernel(const _Float16* __restrict__ q,
                                                   const _Float16* __restrict__ k,
                                                   const _Float16* __restrict__ v,
                                                   _Float16* __restrict__ o) {
  __shared__ __attribute__((aligned(32))) _Float16 Vt[HSs][Tt];      // V transposed: 4 KB
  __shared__ __attribute__((aligned(32))) _Float16 Ps[8][16][Tt];    // 32 KB (wave-private rows)

  int b = blockIdx.x / Hh, h = blockIdx.x % Hh;
  size_t base = (size_t)b * Tt * Cc + h * HSs;
  int tid = threadIdx.x;

  for (int i = tid; i < Tt * HSs; i += 256) {          // transpose on fill
    int t = i >> 4, d = i & 15;
    Vt[d][t] = v[base + (size_t)t * Cc + d];
  }
  __syncthreads();

  int w = tid >> 5, lane = tid & 31, hs = lane >> 4, l16 = lane & 15;

  // --- S = Q K^T (K-dim = HS = 16; A upper half zeroed, so B needs NO zeroing) ---
  FragH qa;
#pragma unroll
  for (int i = 8; i < 16; ++i) qa.e[i] = (_Float16)0.0f;               // k = 16..31 -> 0
  qa.h[0] = *(const v8h*)(q + base + (size_t)(w * 16 + l16) * Cc + hs * 8);  // d = hs*8 + e%8

  v8f S[8];
  v8f zf = {0.f, 0.f, 0.f, 0.f, 0.f, 0.f, 0.f, 0.f};
#pragma unroll
  for (int j = 0; j < 8; ++j) {
    if (j <= w) {                                      // tiles j > w fully masked
      // d = hs*8 + e%8; elements 8..15 feed k>=16, which hits A's zeros
      FragH kb;
      kb.h[0] = *(const v8h*)(k + base + (size_t)(j * 16 + l16) * Cc + hs * 8);
      kb.h[1] = kb.h[0];
      S[j] = __builtin_amdgcn_wmma_f32_16x16x32_f16(false, qa.v, false, kb.v,
                                                    (short)0, zf, false, false);
    }
  }

  // --- causal softmax per query row (row m = r + 8*hs, col = j*16 + l16) ---
  const float scale = 0.25f;  // HS^-0.5
#pragma unroll
  for (int r = 0; r < 8; ++r) {
    int qrow = w * 16 + r + 8 * hs;
    float mx = -1e30f;
#pragma unroll
    for (int j = 0; j < 8; ++j) {
      int col = j * 16 + l16;
      if (col <= qrow) mx = fmaxf(mx, S[j][r] * scale);
    }
#pragma unroll
    for (int m = 8; m >= 1; m >>= 1) mx = fmaxf(mx, __shfl_xor(mx, m, 32));
    float p[8], sum = 0.0f;
#pragma unroll
    for (int j = 0; j < 8; ++j) {
      int col = j * 16 + l16;
      p[j] = (col <= qrow) ? __expf(S[j][r] * scale - mx) : 0.0f;
      sum += p[j];
    }
#pragma unroll
    for (int m = 8; m >= 1; m >>= 1) sum += __shfl_xor(sum, m, 32);
    float inv = 1.0f / sum;
#pragma unroll
    for (int j = 0; j < 8; ++j)                        // write all 128 cols (zeros where masked)
      Ps[w][r + 8 * hs][j * 16 + l16] = (_Float16)(p[j] * inv);
  }
  // Ps is wave-private; same-wave LDS ops are in order — no block barrier needed.

  // --- O = P @ V ; skip K-steps that are entirely past the causal frontier ---
  v8f O = zf;
#pragma unroll
  for (int j = 0; j < 4; ++j) {
    if (j <= (w >> 1)) {
      FragH pa;
      pa.v = *(const v16h*)&Ps[w][l16][j * 32 + hs * 16];
      v16h vb = *(const v16h*)&Vt[l16][j * 32 + hs * 16]; // B: k = hs*16 + e -> consecutive t
      O = __builtin_amdgcn_wmma_f32_16x16x32_f16(false, pa.v, false, vb,
                                                 (short)0, O, false, false);
    }
  }
#pragma unroll
  for (int r = 0; r < 8; ++r)
    o[base + (size_t)(w * 16 + r + 8 * hs) * Cc + l16] = (_Float16)O[r];
}

// ---------------- workspace layout (256B-aligned) ----------------
constexpr size_t OFF_X    = 0;                                   // f32 [M][C]
constexpr size_t OFF_H16  = OFF_X    + (size_t)MM * Cc * 4;      // f16 [M][C]
constexpr size_t OFF_Q16  = OFF_H16  + (size_t)MM * Cc * 2;
constexpr size_t OFF_K16  = OFF_Q16  + (size_t)MM * Cc * 2;
constexpr size_t OFF_V16  = OFF_K16  + (size_t)MM * Cc * 2;
constexpr size_t OFF_O16  = OFF_V16  + (size_t)MM * Cc * 2;
constexpr size_t OFF_FF16 = OFF_O16  + (size_t)MM * Cc * 2;      // f16 [M][FF]
constexpr size_t OFF_WQT  = OFF_FF16 + (size_t)MM * FFf * 2;
constexpr size_t OFF_WKT  = OFF_WQT  + (size_t)Ll * Cc * Cc * 2;
constexpr size_t OFF_WVT  = OFF_WKT  + (size_t)Ll * Cc * Cc * 2;
constexpr size_t OFF_WPT  = OFF_WVT  + (size_t)Ll * Cc * Cc * 2;
constexpr size_t OFF_W1T  = OFF_WPT  + (size_t)Ll * Cc * Cc * 2;
constexpr size_t OFF_W2T  = OFF_W1T  + (size_t)Ll * Cc * FFf * 2;
constexpr size_t OFF_WLMT = OFF_W2T  + (size_t)Ll * Cc * FFf * 2;

extern "C" void kernel_launch(void* const* d_in, const int* in_sizes, int n_in,
                              void* d_out, int out_size, void* d_ws, size_t ws_size,
                              hipStream_t stream) {
  (void)in_sizes; (void)n_in; (void)out_size; (void)ws_size;
  const int*   idx   = (const int*)  d_in[0];
  const float* tok   = (const float*)d_in[1];
  const float* pos   = (const float*)d_in[2];
  const float* Wq    = (const float*)d_in[3];
  const float* Wk    = (const float*)d_in[4];
  const float* Wv    = (const float*)d_in[5];
  const float* Wp    = (const float*)d_in[6];
  const float* bp    = (const float*)d_in[7];
  const float* W1    = (const float*)d_in[8];
  const float* b1    = (const float*)d_in[9];
  const float* W2    = (const float*)d_in[10];
  const float* b2    = (const float*)d_in[11];
  const float* ln1g  = (const float*)d_in[12];
  const float* ln1b  = (const float*)d_in[13];
  const float* ln2g  = (const float*)d_in[14];
  const float* ln2b  = (const float*)d_in[15];
  const float* lnfg  = (const float*)d_in[16];
  const float* lnfb  = (const float*)d_in[17];
  const float* Wlm   = (const float*)d_in[18];
  const float* blm   = (const float*)d_in[19];

  char* W = (char*)d_ws;
  float*     x    = (float*)    (W + OFF_X);
  _Float16*  h16  = (_Float16*) (W + OFF_H16);
  _Float16*  q16  = (_Float16*) (W + OFF_Q16);
  _Float16*  k16  = (_Float16*) (W + OFF_K16);
  _Float16*  v16  = (_Float16*) (W + OFF_V16);
  _Float16*  o16  = (_Float16*) (W + OFF_O16);
  _Float16*  ff16 = (_Float16*) (W + OFF_FF16);
  _Float16*  wqT  = (_Float16*) (W + OFF_WQT);
  _Float16*  wkT  = (_Float16*) (W + OFF_WKT);
  _Float16*  wvT  = (_Float16*) (W + OFF_WVT);
  _Float16*  wpT  = (_Float16*) (W + OFF_WPT);
  _Float16*  w1T  = (_Float16*) (W + OFF_W1T);
  _Float16*  w2T  = (_Float16*) (W + OFF_W2T);
  _Float16*  wlmT = (_Float16*) (W + OFF_WLMT);

  // weight conversion/transposition (exact-cover grids)
  cvt_qkv_kernel<<<216, 256, 0, stream>>>(Wq, wqT);
  cvt_qkv_kernel<<<216, 256, 0, stream>>>(Wk, wkT);
  cvt_qkv_kernel<<<216, 256, 0, stream>>>(Wv, wvT);
  cvt_wp_kernel <<<216, 256, 0, stream>>>(Wp, wpT);
  cvt_w1_kernel <<<864, 256, 0, stream>>>(W1, w1T);
  cvt_w2_kernel <<<864, 256, 0, stream>>>(W2, w2T);
  cvt_lm_kernel <<<30,  256, 0, stream>>>(Wlm, wlmT);

  // x = tok_emb[idx] + pos_emb
  embed_kernel<<<(MM * Cc) / 256, 256, 0, stream>>>(idx, tok, pos, x);

  const int gLN  = MM / 8;                 // 8 tokens (waves) per block
  // strip-mined GEMM grids: waves = (M/16) * ((Npad/16)/NT), blocks = waves/8
  const int g96  = (MM / 16) * ((Cc   / 16) / 6) / 8;   // NT=6 -> 512 blocks
  const int g384 = (MM / 16) * ((FFf  / 16) / 8) / 8;   // NT=8 -> 1536 blocks
  const int gLM  = (MM / 16) * ((VPAD / 16) / 5) / 8;   // NT=5 -> 512 blocks

  for (int l = 0; l < Ll; ++l) {
    // h = LN1(x)
    ln_kernel<<<gLN, 256, 0, stream>>>(x, ln1g + l * Cc, ln1b + l * Cc, h16);
    // q,k,v = h @ W{q,k,v}  (f16 out, heads packed along C)
    gemm_f16_strip<6><<<g96, 256, 0, stream>>>(h16, wqT + (size_t)l * Cc * Cc,
                                               nullptr, nullptr, nullptr, q16, Cc, Cc, Cc, 0);
    gemm_f16_strip<6><<<g96, 256, 0, stream>>>(h16, wkT + (size_t)l * Cc * Cc,
                                               nullptr, nullptr, nullptr, k16, Cc, Cc, Cc, 0);
    gemm_f16_strip<6><<<g96, 256, 0, stream>>>(h16, wvT + (size_t)l * Cc * Cc,
                                               nullptr, nullptr, nullptr, v16, Cc, Cc, Cc, 0);
    // o = softmax(qk^T/4, causal) @ v
    attn_kernel<<<Bb * Hh, 256, 0, stream>>>(q16, k16, v16, o16);
    // x = x + o @ Wp + bp
    gemm_f16_strip<6><<<g96, 256, 0, stream>>>(o16, wpT + (size_t)l * Cc * Cc,
                                               bp + l * Cc, x, x, nullptr, Cc, Cc, Cc, 0);
    // h = LN2(x)
    ln_kernel<<<gLN, 256, 0, stream>>>(x, ln2g + l * Cc, ln2b + l * Cc, h16);
    // ff = relu(h @ W1 + b1)  (f16 out)
    gemm_f16_strip<8><<<g384, 256, 0, stream>>>(h16, w1T + (size_t)l * Cc * FFf,
                                                b1 + l * FFf, nullptr, nullptr, ff16, Cc, FFf, FFf, 1);
    // x = x + ff @ W2 + b2
    gemm_f16_strip<6><<<g96, 256, 0, stream>>>(ff16, w2T + (size_t)l * Cc * FFf,
                                               b2 + l * Cc, x, x, nullptr, FFf, Cc, Cc, 0);
  }

  // final LN + LM head -> logits f32 (N padded 80, stores guarded to 65)
  ln_kernel<<<gLN, 256, 0, stream>>>(x, lnfg, lnfb, h16);
  gemm_f16_strip<5><<<gLM, 256, 0, stream>>>(h16, wlmT, blm, nullptr,
                                             (float*)d_out, nullptr, Cc, VPAD, Vv, 0);
}
